// Attention_local_30949534335056
// MI455X (gfx1250) — matrix operand order repaired and checked by reading, not compiled, not run on hardware
//
#include <hip/hip_runtime.h>
#include <hip/hip_bf16.h>
#include <math.h>

typedef __attribute__((ext_vector_type(16))) _Float16 v16h;
typedef __attribute__((ext_vector_type(8)))  _Float16 v8h;
typedef __attribute__((ext_vector_type(4)))  _Float16 v4h;
typedef __attribute__((ext_vector_type(8)))  float    v8f;
typedef __attribute__((ext_vector_type(4)))  float    v4f;

#define HH     128
#define WW     128
#define DIM    512
#define HEADS  8
#define DHEAD  64
#define TOK    256         // 16x16 tokens per window
#define MWIN   225         // 15x15 candidate windows
#define NW     15
#define KEEP   44
#define BATCH  8
#define NWIN_TOT (BATCH*KEEP)   // 352
#define SCALE  0.125f           // 1/sqrt(64)
#define IOU_T  0.2f

#define LQK 64    // halves per row: q_s / k_s  (256 rows)
#define LVT 256   // halves per row: vt_s (64 rows) and P staging (16 rows/wave)

// ---------------------------------------------------------------- WMMA helpers
__device__ __forceinline__ v8f wmma_f16(v16h a, v16h b, v8f c) {
  return __builtin_amdgcn_wmma_f32_16x16x32_f16(false, a, false, b, (short)0, c,
                                                false, false);
}

// A operand from row-major f16 [16 x K], rstride in halves.
// Lane l<16: K = {k0..k0+7, k0+16..k0+23}; l>=16: +8 (per CDNA5 ISA A layout).
__device__ __forceinline__ v16h ld_a_g(const _Float16* base, int rs, int lane, int k0) {
  int r = lane & 15, hi = lane >> 4;
  const _Float16* p = base + (size_t)r * rs + (k0 + hi * 8);
  v8h lo = *(const v8h*)p;
  v8h hp = *(const v8h*)(p + 16);
  v16h a;
#pragma unroll
  for (int i = 0; i < 8; ++i) { a[i] = lo[i]; a[i + 8] = hp[i]; }
  return a;
}
// B operand (B[k][n] = src[n][k], src row-major N x K):
// lane l holds N=l&15, K = {k0..k0+7, k0+16..k0+23} + 8*(l>>4), matching A.
__device__ __forceinline__ v16h ld_b_g(const _Float16* base, int rs, int lane, int k0) {
  int r = lane & 15, hi = lane >> 4;
  const _Float16* p = base + (size_t)r * rs + (k0 + hi * 8);
  v8h lo = *(const v8h*)p;
  v8h hp = *(const v8h*)(p + 16);
  v16h b;
#pragma unroll
  for (int i = 0; i < 8; ++i) { b[i] = lo[i]; b[i + 8] = hp[i]; }
  return b;
}

// LDS with 16B-chunk XOR swizzle (bank-conflict free 16-row operand gathers).
// mode 0: row stride 64 halves (q/k),  chunk ^= (row>>1)&7
// mode 1: row stride 256 halves (vt/P), chunk ^= row&15
__device__ __forceinline__ int swz_chunk(int row, int chunk, int mode) {
  return mode == 0 ? (chunk ^ ((row >> 1) & 7)) : (chunk ^ (row & 15));
}
__device__ __forceinline__ const v8h* lds_chunk(const _Float16* buf, int rs, int row,
                                                int k, int mode) {
  int c = swz_chunk(row, k >> 3, mode);
  return (const v8h*)(buf + row * rs + (c << 3));
}
__device__ __forceinline__ v16h ld_a_s(const _Float16* buf, int rs, int mode,
                                       int rbase, int lane, int k0) {
  int r = rbase + (lane & 15), hi = lane >> 4;
  v8h lo = *lds_chunk(buf, rs, r, k0 + hi * 8, mode);
  v8h hp = *lds_chunk(buf, rs, r, k0 + hi * 8 + 16, mode);
  v16h a;
#pragma unroll
  for (int i = 0; i < 8; ++i) { a[i] = lo[i]; a[i + 8] = hp[i]; }
  return a;
}
__device__ __forceinline__ v16h ld_b_s(const _Float16* buf, int rs, int mode,
                                       int rbase, int lane, int k0) {
  int r = rbase + (lane & 15), hi = lane >> 4;
  v8h lo = *lds_chunk(buf, rs, r, k0 + hi * 8, mode);
  v8h hp = *lds_chunk(buf, rs, r, k0 + hi * 8 + 16, mode);
  v16h b;
#pragma unroll
  for (int i = 0; i < 8; ++i) { b[i] = lo[i]; b[i + 8] = hp[i]; }
  return b;
}
// Scatter a 16x16 f32 C tile (lane holds col=lane&15, rows j+8*(lane>>4)) into
// a swizzled row-major LDS buffer as f16.
__device__ __forceinline__ void st_c_s(_Float16* buf, int rs, int mode,
                                       int rbase, int cbase, int lane, v8f c) {
  int hi = lane >> 4, cl = cbase + (lane & 15);
#pragma unroll
  for (int j = 0; j < 8; ++j) {
    int row = rbase + j + hi * 8;
    int cc = swz_chunk(row, cl >> 3, mode);
    buf[row * rs + (cc << 3) + (cl & 7)] = (_Float16)c[j];
  }
}
// Transposed scatter (for building V^T from V's C tiles).
__device__ __forceinline__ void st_c_s_T(_Float16* buf, int rs, int mode,
                                         int rbase, int cbase, int lane, v8f c) {
  int hi = lane >> 4;
  int row = cbase + (lane & 15);
#pragma unroll
  for (int j = 0; j < 8; ++j) {
    int cl = rbase + j + hi * 8;
    int cc = swz_chunk(row, cl >> 3, mode);
    buf[row * rs + (cc << 3) + (cl & 7)] = (_Float16)c[j];
  }
}

// ---------------------------------------------------------------- 1. scores
__global__ void k_score(const float* __restrict__ prob, float* __restrict__ score) {
  int bid = blockIdx.x;                 // b*225 + m
  int b = bid / MWIN, m = bid % MWIN;
  int my = m / NW, mx = m % NW;
  int t = threadIdx.x;                  // 64 threads = 8x8 box
  int ey = my * 4 + (t >> 3), ex = mx * 4 + (t & 7);
  float e = 0.f;
#pragma unroll
  for (int c = 0; c < 4; ++c) {
    float p = prob[(((size_t)b * 4 + c) * 64 + ey) * 64 + ex];
    e -= p * log2f(p + 1e-10f);
  }
  __shared__ float red[64];
  red[t] = e;
  __syncthreads();
  if (t == 0) {
    float s = 0.f;
    for (int i = 0; i < 64; ++i) s += red[i];
    score[bid] = s * (1.0f / 64.0f);
  }
}

// ---------------------------------------------------------------- 2. greedy NMS
__global__ void k_nms(const float* __restrict__ score, int* __restrict__ ksy,
                      int* __restrict__ ksx) {
  int b = blockIdx.x, t = threadIdx.x;  // 256 threads
  __shared__ float sc[MWIN];
  __shared__ int ord[MWIN];
  __shared__ int supp[MWIN];
  __shared__ int kept[KEEP];
  if (t < MWIN) sc[t] = score[b * MWIN + t];
  __syncthreads();
  if (t < MWIN) {                       // stable descending rank
    float si = sc[t];
    int r = 0;
    for (int j = 0; j < MWIN; ++j) {
      float sj = sc[j];
      r += (sj > si) || (sj == si && j < t);
    }
    ord[r] = t;
    supp[t] = 0;
  }
  __syncthreads();
  for (int i = 0; i < MWIN - 1; ++i) {
    if (!supp[i] && t > i && t < MWIN) {
      int wa = ord[i], wb = ord[t];
      float ax = (float)((wa % NW) * 8), ay = (float)((wa / NW) * 8);
      float bx = (float)((wb % NW) * 8), by = (float)((wb / NW) * 8);
      float iw = fmaxf(fminf(ax, bx) + 15.f - fmaxf(ax, bx), 0.f);
      float ih = fmaxf(fminf(ay, by) + 15.f - fmaxf(ay, by), 0.f);
      float inter = iw * ih;
      if (inter / (450.f - inter) > IOU_T) supp[t] = 1;
    }
    __syncthreads();
  }
  if (t == 0) {
    int c = 0;
    for (int j = 0; j < MWIN && c < KEEP; ++j)
      if (!supp[j]) kept[c++] = ord[j];
    while (c < KEEP) kept[c++] = ord[MWIN - 1];   // reference padding rule
  }
  __syncthreads();
  if (t < KEEP) {
    int w = kept[t];
    ksy[b * KEEP + t] = (w / NW) * 8;
    ksx[b * KEEP + t] = (w % NW) * 8;
  }
}

// ---------------------------------------------------------------- 3. f32->f16 weights
__global__ void k_cvt(const float* __restrict__ wqkv, const float* __restrict__ wout,
                      _Float16* __restrict__ wqkv_h, _Float16* __restrict__ wout_h) {
  const int n1 = 3 * DIM * DIM, n2 = DIM * DIM;
  for (int i = blockIdx.x * blockDim.x + threadIdx.x; i < n1 + n2;
       i += gridDim.x * blockDim.x) {
    if (i < n1) wqkv_h[i] = (_Float16)wqkv[i];
    else        wout_h[i - n1] = (_Float16)wout[i - n1];
  }
}

// ---------------------------------------------------------------- 4. bilinear patches
__global__ void k_bilin(const float* __restrict__ x, const int* __restrict__ ksy,
                        const int* __restrict__ ksx, _Float16* __restrict__ xi) {
  int win = blockIdx.x;                 // 352
  int b = win / KEEP;
  int t = threadIdx.x;                  // token
  int sy = ksy[win], sx = ksx[win];
  int p = t >> 4, q = t & 15;
  float rp = ((float)p + 0.5f) * 0.9375f;
  float rq = ((float)q + 0.5f) * 0.9375f;
  int i0 = (int)floorf(rp), j0 = (int)floorf(rq);
  float fp = rp - (float)i0, fq = rq - (float)j0;
  const float* p00 = x + (((size_t)b * (HH * WW)) + (size_t)(sy + i0) * WW + (sx + j0)) * DIM;
  const float* p01 = p00 + DIM;
  const float* p10 = p00 + (size_t)WW * DIM;
  const float* p11 = p10 + DIM;
  float w00 = (1.f - fp) * (1.f - fq), w01 = (1.f - fp) * fq;
  float w10 = fp * (1.f - fq),         w11 = fp * fq;
  _Float16* dst = xi + ((size_t)win * TOK + t) * DIM;
  for (int d = 0; d < DIM; d += 4) {
    v4f a = *(const v4f*)(p00 + d), c = *(const v4f*)(p01 + d);
    v4f e = *(const v4f*)(p10 + d), f = *(const v4f*)(p11 + d);
    v4h o;
#pragma unroll
    for (int i = 0; i < 4; ++i)
      o[i] = (_Float16)(w00 * a[i] + w01 * c[i] + w10 * e[i] + w11 * f[i]);
    *(v4h*)(dst + d) = o;
  }
}

// ---------------------------------------------------------------- 5. per-window attention
__global__ void __launch_bounds__(256)
k_attn(const _Float16* __restrict__ xi, const _Float16* __restrict__ wqkv,
       _Float16* __restrict__ og) {
  extern __shared__ _Float16 lds[];
  _Float16* q_s  = lds;                       // 256 x 64
  _Float16* k_s  = q_s + TOK * LQK;           // 256 x 64
  _Float16* vt_s = k_s + TOK * LQK;           // 64 x 256  (V transposed)
  _Float16* p_s  = vt_s + DHEAD * LVT;        // 8 waves x 16 x 256
  int win = blockIdx.x;
  int wave = threadIdx.x >> 5, lane = threadIdx.x & 31, hi = lane >> 4;
  const _Float16* xw = xi + (size_t)win * TOK * DIM;
  _Float16* ow = og + (size_t)win * TOK * DIM;
  _Float16* pw = p_s + wave * 16 * LVT;

  for (int h = 0; h < HEADS; ++h) {
    // ---- q/k/v head projections: 3 x (256x64) = 192 tiles over 8 waves ----
    for (int tt = wave; tt < 192; tt += 8) {
      int mat = tt / 64, rem = tt % 64, mt = rem >> 2, nt = rem & 3;
      const _Float16* wb = wqkv + ((size_t)(mat * DIM + h * DHEAD + nt * 16)) * DIM;
      const _Float16* ab = xw + (size_t)(mt * 16) * DIM;
      v8f c = {};
#pragma unroll 4
      for (int ks = 0; ks < 16; ++ks) {
        v16h a = ld_a_g(ab, DIM, lane, ks * 32);
        v16h b = ld_b_g(wb, DIM, lane, ks * 32);
        c = wmma_f16(a, b, c);
      }
      if (mat == 0)      st_c_s(q_s, LQK, 0, mt * 16, nt * 16, lane, c);
      else if (mat == 1) st_c_s(k_s, LQK, 0, mt * 16, nt * 16, lane, c);
      else               st_c_s_T(vt_s, LVT, 1, mt * 16, nt * 16, lane, c);
    }
    __syncthreads();

    // ---- attention: each wave owns 2 query tiles of 16 rows ----
    for (int mt = wave; mt < 16; mt += 8) {
      v16h qa0 = ld_a_s(q_s, LQK, 0, mt * 16, lane, 0);
      v16h qa1 = ld_a_s(q_s, LQK, 0, mt * 16, lane, 32);
      v8f s[16];
#pragma unroll
      for (int nt = 0; nt < 16; ++nt) {
        v16h b0 = ld_b_s(k_s, LQK, 0, nt * 16, lane, 0);
        v16h b1 = ld_b_s(k_s, LQK, 0, nt * 16, lane, 32);
        v8f c = {};
        c = wmma_f16(qa0, b0, c);
        c = wmma_f16(qa1, b1, c);
        s[nt] = c;
      }
      // softmax over rows (row m lives in 16 lanes of one half-wave)
      float mx[8], sm[8];
#pragma unroll
      for (int j = 0; j < 8; ++j) mx[j] = -3.0e38f;
#pragma unroll
      for (int nt = 0; nt < 16; ++nt)
#pragma unroll
        for (int j = 0; j < 8; ++j) mx[j] = fmaxf(mx[j], s[nt][j]);
#pragma unroll
      for (int j = 0; j < 8; ++j) {
#pragma unroll
        for (int off = 1; off < 16; off <<= 1)
          mx[j] = fmaxf(mx[j], __shfl_xor(mx[j], off, 32));
        sm[j] = 0.f;
      }
#pragma unroll
      for (int nt = 0; nt < 16; ++nt)
#pragma unroll
        for (int j = 0; j < 8; ++j) {
          float e = __expf((s[nt][j] - mx[j]) * SCALE);
          s[nt][j] = e;
          sm[j] += e;
        }
#pragma unroll
      for (int j = 0; j < 8; ++j) {
#pragma unroll
        for (int off = 1; off < 16; off <<= 1)
          sm[j] += __shfl_xor(sm[j], off, 32);
        sm[j] = 1.0f / sm[j];
      }
      // stage P (16x256 f16) per-wave so it re-enters WMMA as an A operand
#pragma unroll
      for (int nt = 0; nt < 16; ++nt) {
        v8f pv;
#pragma unroll
        for (int j = 0; j < 8; ++j) pv[j] = s[nt][j] * sm[j];
        st_c_s(pw, LVT, 1, 0, nt * 16, lane, pv);
      }
      // O = P @ V  (via V^T rows as B operand)
      v8f o[4] = {};
#pragma unroll
      for (int ks = 0; ks < 8; ++ks) {
        v16h pa = ld_a_s(pw, LVT, 1, 0, lane, ks * 32);
#pragma unroll
        for (int nto = 0; nto < 4; ++nto) {
          v16h vb = ld_b_s(vt_s, LVT, 1, nto * 16, lane, ks * 32);
          o[nto] = wmma_f16(pa, vb, o[nto]);
        }
      }
#pragma unroll
      for (int nto = 0; nto < 4; ++nto)
#pragma unroll
        for (int j = 0; j < 8; ++j) {
          int m = mt * 16 + j + hi * 8;
          int col = h * DHEAD + nto * 16 + (lane & 15);
          ow[(size_t)m * DIM + col] = (_Float16)o[nto][j];
        }
    }
    __syncthreads();
  }
}

// ---------------------------------------------------------------- 6. zero acc+cnt
__global__ void k_zero(float* __restrict__ p, size_t n4) {
  size_t stride = (size_t)gridDim.x * blockDim.x;
  v4f z = {};
  for (size_t i = (size_t)blockIdx.x * blockDim.x + threadIdx.x; i < n4; i += stride)
    *((v4f*)p + i) = z;
}

// ---------------------------------------------------------------- 7. out-proj + scatter
__global__ void __launch_bounds__(256)
k_proj(const _Float16* __restrict__ og, const _Float16* __restrict__ wout,
       const float* __restrict__ bout, const int* __restrict__ ksy,
       const int* __restrict__ ksx, float* __restrict__ acc, float* __restrict__ cnt) {
  int win = blockIdx.x;
  int b = win / KEEP;
  int sy = ksy[win], sx = ksx[win];
  int wave = threadIdx.x >> 5, lane = threadIdx.x & 31, hi = lane >> 4;
  {
    int t = threadIdx.x;                // coverage counts
    int y = sy + (t >> 4), xx = sx + (t & 15);
    atomicAdd(&cnt[(size_t)b * (HH * WW) + (size_t)y * WW + xx], 1.0f);
  }
  const _Float16* ob = og + (size_t)win * TOK * DIM;
  for (int tt = wave; tt < 512; tt += 8) {  // 16 mtiles x 32 ntiles
    int mt = tt >> 5, nt = tt & 31;
    const _Float16* ab = ob + (size_t)(mt * 16) * DIM;
    const _Float16* bb = wout + (size_t)(nt * 16) * DIM;
    v8f c = {};
#pragma unroll 4
    for (int ks = 0; ks < 16; ++ks) {
      v16h a  = ld_a_g(ab, DIM, lane, ks * 32);
      v16h b2 = ld_b_g(bb, DIM, lane, ks * 32);
      c = wmma_f16(a, b2, c);
    }
    int col = nt * 16 + (lane & 15);
    float bias = bout[col];
#pragma unroll
    for (int j = 0; j < 8; ++j) {
      int m = mt * 16 + j + hi * 8;
      int y = sy + (m >> 4), xx = sx + (m & 15);
      atomicAdd(&acc[((size_t)b * (HH * WW) + (size_t)y * WW + xx) * DIM + col],
                c[j] + bias);
    }
  }
}

// ---------------------------------------------------------------- 8. final combine
__global__ void k_final(const float* __restrict__ x, const float* __restrict__ acc,
                        const float* __restrict__ cnt, float* __restrict__ out) {
  size_t n4 = (size_t)BATCH * HH * WW * DIM / 4;
  size_t stride = (size_t)gridDim.x * blockDim.x;
  for (size_t i = (size_t)blockIdx.x * blockDim.x + threadIdx.x; i < n4; i += stride) {
    size_t pix = (i * 4) / DIM;
    float inv = 1.0f / (cnt[pix] + 1e-10f);
    v4f xv = *((const v4f*)x + i);
    v4f av = *((const v4f*)acc + i);
    v4f o;
#pragma unroll
    for (int k = 0; k < 4; ++k) o[k] = xv[k] + av[k] * inv;
    *((v4f*)out + i) = o;
  }
}

// ---------------------------------------------------------------- launch
extern "C" void kernel_launch(void* const* d_in, const int* in_sizes, int n_in,
                              void* d_out, int out_size, void* d_ws, size_t ws_size,
                              hipStream_t stream) {
  (void)in_sizes; (void)n_in; (void)out_size; (void)ws_size;
  const float* x    = (const float*)d_in[0];
  const float* prob = (const float*)d_in[1];
  // d_in[2] = W_fix (all-ones 8x8 box) folded into k_score's mean.
  const float* wqkv = (const float*)d_in[3];
  const float* wout = (const float*)d_in[4];
  const float* bout = (const float*)d_in[5];
  float* out = (float*)d_out;

  char* ws = (char*)d_ws;
  size_t cur = 0;
  auto take = [&](size_t bytes) -> char* {
    char* p = ws + cur;
    cur += (bytes + 255) & ~(size_t)255;
    return p;
  };
  float*    score  = (float*)take((size_t)BATCH * MWIN * sizeof(float));
  int*      ksy    = (int*)take((size_t)NWIN_TOT * sizeof(int));
  int*      ksx    = (int*)take((size_t)NWIN_TOT * sizeof(int));
  _Float16* wqkv_h = (_Float16*)take((size_t)3 * DIM * DIM * 2);
  _Float16* wout_h = (_Float16*)take((size_t)DIM * DIM * 2);
  _Float16* og     = (_Float16*)take((size_t)NWIN_TOT * TOK * DIM * 2);
  float*    acc    = (float*)take((size_t)BATCH * HH * WW * DIM * sizeof(float));
  float*    cnt    = (float*)take((size_t)BATCH * HH * WW * sizeof(float));
  _Float16* xi     = (_Float16*)acc;   // alias: xi is dead before acc is zeroed

  k_score<<<BATCH * MWIN, 64, 0, stream>>>(prob, score);
  k_nms<<<BATCH, 256, 0, stream>>>(score, ksy, ksx);
  k_cvt<<<1024, 256, 0, stream>>>(wqkv, wout, wqkv_h, wout_h);
  k_bilin<<<NWIN_TOT, 256, 0, stream>>>(x, ksy, ksx, xi);

  size_t lds_bytes = (size_t)(2 * TOK * LQK + DHEAD * LVT + 8 * 16 * LVT) * sizeof(_Float16); // 160 KB
  k_attn<<<NWIN_TOT, 256, lds_bytes, stream>>>(xi, wqkv_h, og);

  size_t zero4 = ((size_t)BATCH * HH * WW * DIM * 4 + (size_t)BATCH * HH * WW * 4) / 16;
  k_zero<<<4096, 256, 0, stream>>>(acc, zero4);   // acc and cnt are contiguous
  k_proj<<<NWIN_TOT, 256, 0, stream>>>(og, wout_h, bout, ksy, ksx, acc, cnt);
  k_final<<<8192, 256, 0, stream>>>(x, acc, cnt, out);
}
